// ResiduePairEncoder_17935783428596
// MI455X (gfx1250) — matrix operand order repaired and checked by reading, not compiled, not run on hardware
//
#include <hip/hip_runtime.h>
#include <math.h>

// ---------------- constants ----------------
#define NB 2
#define NN 1536
#define K1 32
#define K2 16
#define K3 9
#define KTOT 57
#define NEDGE (NB*NN*KTOT)      // 175104
#define KF 462
#define KPAD 480                 // 15 K-tiles of 32
#define DO 128
#define EPSD 1e-6f
#define LRHALF 32
#define SEQHALF 4

typedef __attribute__((ext_vector_type(16))) _Float16 v16h;
typedef __attribute__((ext_vector_type(8)))  float    v8f;

// atom order in ws_atoms: N=0, Ca=1, C=2, O=3, Cb=4  (3 floats each, 15 per residue)
__device__ __constant__ int PAIR_A[26] = {1,0,2,3,4, 1,1,1,1, 0,0,0, 4,4, 3, 0,2,3,4, 2,3,4, 2,3, 2, 4};
__device__ __constant__ int PAIR_B[26] = {1,0,2,3,4, 0,2,3,4, 2,3,4, 2,3, 2, 1,1,1,1, 0,0,0, 4,4, 3, 4};

__device__ __forceinline__ float sgnf(float x) { return (x > 0.f) ? 1.f : ((x < 0.f) ? -1.f : 0.f); }

__device__ __forceinline__ void norm3(float& x, float& y, float& z) {
    float inv = 1.0f / sqrtf(x*x + y*y + z*z + 1e-12f);
    x *= inv; y *= inv; z *= inv;
}

// ---------------- kernel 1: atoms + virtual Cb ----------------
__global__ __launch_bounds__(256) void geom_atoms_kernel(
    const float* __restrict__ pos, const unsigned char* __restrict__ mask_atoms,
    float* __restrict__ atoms)
{
    int t = blockIdx.x * blockDim.x + threadIdx.x;
    if (t >= NB*NN) return;
    const float* p = pos + (long)t * 15 * 3;
    float Nx=p[0],Ny=p[1],Nz=p[2];
    float Cax=p[3],Cay=p[4],Caz=p[5];
    float Cx=p[6],Cy=p[7],Cz=p[8];
    float Ox=p[9],Oy=p[10],Oz=p[11];
    float bx=Cax-Nx, by=Cay-Ny, bz=Caz-Nz;
    float cx=Cx-Cax, cy=Cy-Cay, cz=Cz-Caz;
    float ax=by*cz-bz*cy, ay=bz*cx-bx*cz, az=bx*cy-by*cx;
    float vx = -0.58273431f*ax + 0.56802827f*bx - 0.54067466f*cx + Cax;
    float vy = -0.58273431f*ay + 0.56802827f*by - 0.54067466f*cy + Cay;
    float vz = -0.58273431f*az + 0.56802827f*bz - 0.54067466f*cz + Caz;
    float Cbx, Cby, Cbz;
    if (mask_atoms[(long)t*15 + 4]) { Cbx=p[12]; Cby=p[13]; Cbz=p[14]; }
    else { Cbx=vx; Cby=vy; Cbz=vz; }
    float* o = atoms + (long)t * 15;
    o[0]=Nx;  o[1]=Ny;  o[2]=Nz;
    o[3]=Cax; o[4]=Cay; o[5]=Caz;
    o[6]=Cx;  o[7]=Cy;  o[8]=Cz;
    o[9]=Ox;  o[10]=Oy; o[11]=Oz;
    o[12]=Cbx;o[13]=Cby;o[14]=Cbz;
}

// ---------------- kernel 2: local frames ----------------
__global__ __launch_bounds__(256) void geom_frames_kernel(
    const float* __restrict__ atoms, float* __restrict__ frames)
{
    int t = blockIdx.x * blockDim.x + threadIdx.x;
    if (t >= NB*NN) return;
    int n = t % NN;
    float ux, uy, uz;
    if (n == 0) { ux=uy=uz=1.0f; norm3(ux,uy,uz); }
    else {
        ux = atoms[(long)t*15+12] - atoms[(long)(t-1)*15+12];
        uy = atoms[(long)t*15+13] - atoms[(long)(t-1)*15+13];
        uz = atoms[(long)t*15+14] - atoms[(long)(t-1)*15+14];
        norm3(ux,uy,uz);
    }
    float bx,by,bz, nx,ny,nz;
    if (n == NN-1) {
        bx=by=bz=1.0f; norm3(bx,by,bz);
        nx=ny=nz=1.0f; norm3(nx,ny,nz);
    } else {
        float vx = atoms[(long)(t+1)*15+12] - atoms[(long)t*15+12];
        float vy = atoms[(long)(t+1)*15+13] - atoms[(long)t*15+13];
        float vz = atoms[(long)(t+1)*15+14] - atoms[(long)t*15+14];
        norm3(vx,vy,vz);
        bx = ux - vx; by = uy - vy; bz = uz - vz; norm3(bx,by,bz);
        nx = uy*vz - uz*vy; ny = uz*vx - ux*vz; nz = ux*vy - uy*vx; norm3(nx,ny,nz);
    }
    float cx = by*nz - bz*ny, cy = bz*nx - bx*nz, cz = bx*ny - by*nx;  // cross(bvec,nvec)
    float* f = frames + (long)t * 9;
    f[0]=bx; f[1]=by; f[2]=bz;   // row 0 = bvec
    f[3]=nx; f[4]=ny; f[5]=nz;   // row 1 = nvec
    f[6]=cx; f[7]=cy; f[8]=cz;   // row 2 = cross
}

// ---------------- kernel 3: edge_w -> fp16, pre-swizzled into WMMA B-frag layout ----
// Bsw[((kt*8 + wcol)*32 + lane)*8 + v] packs B[K=kt*32+16*hi+2v][col] and B[K+1][col]
__global__ __launch_bounds__(256) void convert_w_kernel(
    const float* __restrict__ edge_w, unsigned int* __restrict__ bsw)
{
    int idx = blockIdx.x * blockDim.x + threadIdx.x;
    if (idx >= 15*8*32*8) return;
    int v    = idx & 7;
    int lane = (idx >> 3) & 31;
    int wcol = (idx >> 8) & 7;
    int kt   = idx >> 11;
    int n  = lane & 15, hi = lane >> 4;
    int col = wcol*16 + n;
    int k0 = kt*32 + 16*hi + 2*v;
    float f0 = (k0   < KF) ? edge_w[(long)k0*DO + col]     : 0.0f;
    float f1 = (k0+1 < KF) ? edge_w[(long)(k0+1)*DO + col] : 0.0f;
    union { _Float16 h[2]; unsigned int u; } pk;
    pk.h[0] = (_Float16)f0; pk.h[1] = (_Float16)f1;
    bsw[idx] = pk.u;
}

// ---------------- kernel 4: per-row top-k (spatial/long-range/sequential) --------
__global__ __launch_bounds__(256) void topk_kernel(
    const float* __restrict__ atoms, const unsigned char* __restrict__ mask,
    const int* __restrict__ ridx, int* __restrict__ eidx_ws,
    int* __restrict__ out_idx, float* __restrict__ out_ma)
{
    __shared__ float Drow[NN];
    __shared__ float Dwork[NN];
    __shared__ float redv[256];
    __shared__ int   redi[256];
    __shared__ float s_dmax;

    const int tid = threadIdx.x;
    const int bi  = blockIdx.x;          // b*NN + i
    const int b   = bi / NN;
    const int mi  = mask[bi] ? 1 : 0;
    const int ri  = ridx[bi];
    const float cx = atoms[(long)bi*15+12];
    const float cy = atoms[(long)bi*15+13];
    const float cz = atoms[(long)bi*15+14];

    float lmax = 0.0f;
    for (int j = tid; j < NN; j += 256) {
        long gj = (long)b*NN + j;
        float dx = cx - atoms[gj*15+12];
        float dy = cy - atoms[gj*15+13];
        float dz = cz - atoms[gj*15+14];
        int m2 = mi & (mask[gj] ? 1 : 0);
        float D = m2 ? sqrtf(dx*dx + dy*dy + dz*dz + EPSD) : 0.0f;
        Drow[j] = D;
        lmax = fmaxf(lmax, D);
    }
    redv[tid] = lmax; __syncthreads();
    for (int s = 128; s > 0; s >>= 1) {
        if (tid < s) redv[tid] = fmaxf(redv[tid], redv[tid+s]);
        __syncthreads();
    }
    if (tid == 0) s_dmax = redv[0];
    __syncthreads();
    const float dmax = s_dmax;

    // -------- phase helper (inlined 3x) --------
    for (int phase = 0; phase < 3; ++phase) {
        int kcount = (phase==0) ? K1 : (phase==1 ? K2 : K3);
        int koff   = (phase==0) ? 0  : (phase==1 ? K1 : K1+K2);
        int argmax = (phase == 2) ? 1 : 0;
        for (int j = tid; j < NN; j += 256) {
            int m2 = mi & (mask[(long)b*NN + j] ? 1 : 0);
            int rel = ri - ridx[(long)b*NN + j];
            int ar  = rel < 0 ? -rel : rel;
            float val;
            if (phase == 0)      val = Drow[j] + (m2 ? 0.0f : dmax);
            else if (phase == 1) val = Drow[j] + (((ar > LRHALF) & m2) ? 0.0f : dmax);
            else                 val = ((ar <= SEQHALF) & m2) ? Drow[j] : 0.0f;
            Dwork[j] = val;
        }
        __syncthreads();
        for (int sel = 0; sel < kcount; ++sel) {
            float bv = argmax ? -3.0e38f : 3.0e38f;
            int   bj = NN;
            for (int j = tid; j < NN; j += 256) {
                float v = Dwork[j];
                bool better = argmax ? (v > bv || (v == bv && j < bj))
                                     : (v < bv || (v == bv && j < bj));
                if (better) { bv = v; bj = j; }
            }
            redv[tid] = bv; redi[tid] = bj; __syncthreads();
            for (int s = 128; s > 0; s >>= 1) {
                if (tid < s) {
                    float v2 = redv[tid+s]; int i2 = redi[tid+s];
                    bool better = argmax ? (v2 > redv[tid] || (v2 == redv[tid] && i2 < redi[tid]))
                                         : (v2 < redv[tid] || (v2 == redv[tid] && i2 < redi[tid]));
                    if (better) { redv[tid] = v2; redi[tid] = i2; }
                }
                __syncthreads();
            }
            if (tid == 0) {
                int js = redi[0];
                Dwork[js] = argmax ? -3.0e38f : 3.0e38f;
                long off = (long)bi*KTOT + koff + sel;
                int mj = mask[(long)b*NN + js] ? 1 : 0;
                int m2 = mi & mj;
                int rel = ri - ridx[(long)b*NN + js];
                int ar  = rel < 0 ? -rel : rel;
                int gm;
                if (phase == 0)      gm = m2;
                else if (phase == 1) gm = (ar > LRHALF) & m2;
                else                 gm = (ar <= SEQHALF) & m2;
                eidx_ws[off] = js;
                out_idx[off] = js;
                out_ma[off]  = (float)(mi & gm);
            }
            __syncthreads();
        }
    }
}

// ---------------- kernel 5: fused features + WMMA GEMM + LayerNorm ---------------
__global__ __launch_bounds__(256) void fused_feat_gemm_ln_kernel(
    const float* __restrict__ atoms, const float* __restrict__ frames,
    const unsigned int* __restrict__ bsw, const int* __restrict__ eidx,
    const float* __restrict__ relpos_w, const float* __restrict__ chains_w,
    const float* __restrict__ aapair_w,
    const int* __restrict__ ridx, const int* __restrict__ chain_nb,
    const int* __restrict__ aa, const unsigned char* __restrict__ mask,
    float* __restrict__ out_h)
{
    __shared__ _Float16 Esh[16*KPAD];       // 16 edges x 480 fp16 features
    __shared__ float s_ed[16][48];          // atoms_i(15) atoms_j(15) frame_i(9) frame_j(9)
    __shared__ float s_dist[16][26];
    __shared__ float s_tq[16][8];           // t0..2, q0..3
    __shared__ float s_h[16][DO];
    __shared__ float s_mu[16], s_rs[16];

    const int tid = threadIdx.x;
    const int er  = tid >> 4;               // local edge 0..15
    const int j16 = tid & 15;               // slice within edge
    const long e  = (long)blockIdx.x*16 + er;
    const int b   = (int)(e / ((long)NN*KTOT));
    const int rr  = (int)(e % ((long)NN*KTOT));
    const int i   = rr / KTOT;
    const int k   = rr % KTOT;
    const int bi  = b*NN + i;
    const int j   = eidx[e];
    const int bj  = b*NN + j;

    // cooperative load of per-edge geometry (48 floats by 16 threads)
    #pragma unroll
    for (int t = 0; t < 3; ++t) {
        int idx = j16*3 + t;
        float v;
        if (idx < 15)      v = atoms[(long)bi*15 + idx];
        else if (idx < 30) v = atoms[(long)bj*15 + (idx-15)];
        else if (idx < 39) v = frames[(long)bi*9 + (idx-30)];
        else               v = frames[(long)bj*9 + (idx-39)];
        s_ed[er][idx] = v;
    }
    __syncthreads();

    // per-thread scalars (redundant across the 16 threads of an edge)
    int rel  = ridx[bi] - ridx[bj];
    int ceq  = (chain_nb[bi] == chain_nb[bj]) ? 1 : 0;
    int mi   = mask[bi] ? 1 : 0;
    int mj   = mask[bj] ? 1 : 0;
    int m2   = mi & mj;
    int ar   = rel < 0 ? -rel : rel;
    int gm;
    if (k < K1)          gm = m2;
    else if (k < K1+K2)  gm = (ar > LRHALF) & m2;
    else                 gm = (ar <= SEQHALF) & m2;
    float ma = (float)(mi & gm);
    int dclip = rel + 32; dclip = dclip < 0 ? 0 : (dclip > 64 ? 64 : dclip);
    int ai2 = (aa[bi] + 1) % 22, aj2 = (aa[bj] + 1) % 22;
    int pair = ai2*22 + aj2;
    if (pair < 21) pair = 21;
    if (pair % 22 == 0) pair = 21;

    // 26 inter-atom distances (shared within the edge)
    for (int p = j16; p < 26; p += 16) {
        int an = PAIR_A[p], bn = PAIR_B[p];
        float dx = s_ed[er][an*3+0] - s_ed[er][15 + bn*3+0];
        float dy = s_ed[er][an*3+1] - s_ed[er][15 + bn*3+1];
        float dz = s_ed[er][an*3+2] - s_ed[er][15 + bn*3+2];
        s_dist[er][p] = sqrtf(dx*dx + dy*dy + dz*dz + EPSD);
    }
    // orientation features: one thread per edge
    if (j16 == 0) {
        const float* Fi = &s_ed[er][30];
        const float* Fj = &s_ed[er][39];
        float dX0 = s_ed[er][15+12] - s_ed[er][12];
        float dX1 = s_ed[er][15+13] - s_ed[er][13];
        float dX2 = s_ed[er][15+14] - s_ed[er][14];
        float t0 = Fi[0]*dX0 + Fi[1]*dX1 + Fi[2]*dX2;
        float t1 = Fi[3]*dX0 + Fi[4]*dX1 + Fi[5]*dX2;
        float t2 = Fi[6]*dX0 + Fi[7]*dX1 + Fi[8]*dX2;
        float ti = 1.0f / sqrtf(t0*t0 + t1*t1 + t2*t2 + 1e-12f);
        s_tq[er][0] = t0*ti*ma; s_tq[er][1] = t1*ti*ma; s_tq[er][2] = t2*ti*ma;
        // r = frame_i^T @ frame_j  : r[a][l] = sum_c Fi[c][a]*Fj[c][l]
        float r[3][3];
        #pragma unroll
        for (int a3 = 0; a3 < 3; ++a3)
            #pragma unroll
            for (int l = 0; l < 3; ++l)
                r[a3][l] = Fi[0*3+a3]*Fj[0*3+l] + Fi[1*3+a3]*Fj[1*3+l] + Fi[2*3+a3]*Fj[2*3+l];
        float Rxx = r[0][0], Ryy = r[1][1], Rzz = r[2][2];
        float m0 = 0.5f*sqrtf(fabsf(1.f + Rxx - Ryy - Rzz) + 1e-8f);
        float m1 = 0.5f*sqrtf(fabsf(1.f - Rxx + Ryy - Rzz) + 1e-8f);
        float m2q= 0.5f*sqrtf(fabsf(1.f - Rxx - Ryy + Rzz) + 1e-8f);
        float qx = sgnf(r[2][1]-r[1][2])*m0;
        float qy = sgnf(r[0][2]-r[2][0])*m1;
        float qz = sgnf(r[1][0]-r[0][1])*m2q;
        float tr = 1.f + Rxx + Ryy + Rzz;
        float qw = sqrtf((tr > 0.f ? tr : 0.f) + 1e-8f)*0.5f;
        float qi = 1.0f / sqrtf(qx*qx + qy*qy + qz*qz + qw*qw + 1e-12f);
        s_tq[er][3] = qx*qi*ma; s_tq[er][4] = qy*qi*ma;
        s_tq[er][5] = qz*qi*ma; s_tq[er][6] = qw*qi*ma;
    }
    __syncthreads();

    // build the 462(+pad) feature vector in fp16
    for (int f = j16; f < KPAD; f += 16) {
        float v;
        if (f < 16) {
            v = relpos_w[dclip*16 + f] + chains_w[ceq*16 + f];
        } else if (f < 432) {
            int p = (f - 16) >> 4, c = (f - 16) & 15;
            float mu = 2.0f + (20.0f/15.0f)*(float)c;
            float x  = (s_dist[er][p] - mu) * 0.8f;          // sigma = 1.25
            v = __expf(-x*x);
        } else if (f < 446) {
            int o = f - 432;
            if (o < 7)       v = s_tq[er][o];                 // t(3), q(4) already masked
            else             v = (1.0f - 2.0f*s_tq[er][o-7])*ma;  // t2(3), r2(4)
        } else if (f < 462) {
            v = aapair_w[pair*16 + (f - 446)] * ma;
        } else {
            v = 0.0f;                                         // K padding
        }
        Esh[er*KPAD + f] = (_Float16)v;
    }
    __syncthreads();

    // -------- WMMA GEMM: 16x480 (LDS) x 480x128 (pre-swizzled global) --------
    const int lane = tid & 31;
    const int wv   = tid >> 5;        // wave -> column tile 0..7
    const int M    = lane & 15;
    const int hi   = lane >> 4;
    v8f acc = {0.f,0.f,0.f,0.f,0.f,0.f,0.f,0.f};
    const unsigned int* Eu = (const unsigned int*)Esh;   // 240 dwords per row
    union { v16h v; unsigned int u[8]; } af, bf;
    for (int kt = 0; kt < 15; ++kt) {
        #pragma unroll
        for (int v2 = 0; v2 < 8; ++v2) {
            int kbase = 2*v2 + 16*hi; // contiguous k order, matches Bsw
            af.u[v2] = Eu[M*240 + ((kt*32 + kbase) >> 1)];
        }
        const uint4* Bp = (const uint4*)bsw + ((size_t)(kt*8 + wv)*32 + lane)*2;
        uint4 p0 = Bp[0], p1 = Bp[1];
        bf.u[0]=p0.x; bf.u[1]=p0.y; bf.u[2]=p0.z; bf.u[3]=p0.w;
        bf.u[4]=p1.x; bf.u[5]=p1.y; bf.u[6]=p1.z; bf.u[7]=p1.w;
        acc = __builtin_amdgcn_wmma_f32_16x16x32_f16(false, af.v, false, bf.v,
                                                     (short)0, acc, false, false);
    }
    #pragma unroll
    for (int r2 = 0; r2 < 8; ++r2)
        s_h[r2 + 8*hi][wv*16 + M] = acc[r2];
    __syncthreads();

    // -------- LayerNorm over 128 --------
    if (tid < 16) {
        float s = 0.f, ss = 0.f;
        for (int c = 0; c < DO; ++c) { float x = s_h[tid][c]; s += x; ss += x*x; }
        float mu = s * (1.0f/DO);
        float var = ss * (1.0f/DO) - mu*mu;
        s_mu[tid] = mu;
        s_rs[tid] = rsqrtf(var + 1e-5f);
    }
    __syncthreads();
    #pragma unroll
    for (int t = 0; t < 8; ++t) {
        int idx = tid*8 + t;           // 0..2047
        int m   = idx >> 7, col = idx & 127;
        long eo = (long)blockIdx.x*16 + m;
        out_h[eo*DO + col] = (s_h[m][col] - s_mu[m]) * s_rs[m];
    }
}

// ---------------- launcher ----------------
extern "C" void kernel_launch(void* const* d_in, const int* in_sizes, int n_in,
                              void* d_out, int out_size, void* d_ws, size_t ws_size,
                              hipStream_t stream) {
    const float* pos        = (const float*)d_in[0];
    const float* relpos_w   = (const float*)d_in[1];
    const float* chains_w   = (const float*)d_in[2];
    const float* aapair_w   = (const float*)d_in[3];
    const float* edge_w     = (const float*)d_in[4];
    const unsigned char* mask_atoms = (const unsigned char*)d_in[5];
    const int*   residue_idx= (const int*)d_in[6];
    const int*   chain_nb   = (const int*)d_in[7];
    const int*   aa         = (const int*)d_in[8];
    const unsigned char* mask = (const unsigned char*)d_in[9];

    char* ws = (char*)d_ws;
    float*        ws_atoms  = (float*)ws;                       // 46080 f32
    float*        ws_frames = (float*)(ws + 184320);            // 27648 f32
    unsigned int* ws_bsw    = (unsigned int*)(ws + 294912);     // 30720 u32 (16B aligned)
    int*          ws_eidx   = (int*)(ws + 417792);              // 175104 i32

    float* out_h   = (float*)d_out;                             // 22413312 f32
    int*   out_idx = (int*)(out_h + (long)NEDGE*DO);            // 175104 i32 (bit-stored)
    float* out_ma  = (float*)(out_h + (long)NEDGE*DO) + NEDGE;  // 175104 f32

    geom_atoms_kernel<<<(NB*NN + 255)/256, 256, 0, stream>>>(pos, mask_atoms, ws_atoms);
    geom_frames_kernel<<<(NB*NN + 255)/256, 256, 0, stream>>>(ws_atoms, ws_frames);
    convert_w_kernel<<<(15*8*32*8 + 255)/256, 256, 0, stream>>>(edge_w, ws_bsw);
    topk_kernel<<<NB*NN, 256, 0, stream>>>(ws_atoms, mask, residue_idx,
                                           ws_eidx, out_idx, out_ma);
    fused_feat_gemm_ln_kernel<<<NEDGE/16, 256, 0, stream>>>(
        ws_atoms, ws_frames, ws_bsw, ws_eidx,
        relpos_w, chains_w, aapair_w,
        residue_idx, chain_nb, aa, mask, out_h);
}
